// AttentionalReadout_40046275068524
// MI455X (gfx1250) — matrix operand order repaired and checked by reading, not compiled, not run on hardware
//
#include <hip/hip_runtime.h>
#include <hip/hip_bf16.h>
#include <math.h>
#include <stdint.h>

#define HIDDEN 256
#define HALF_H 128
#define NUM_GRAPHS 8192
#define NPB 2048  // nodes per block in readout

typedef __attribute__((ext_vector_type(16))) __bf16 v16bf;
typedef __attribute__((ext_vector_type(8)))  float  v8f;

__device__ __forceinline__ unsigned short f32_to_bf16(float f) {
    unsigned u = __float_as_uint(f);
    u += 0x7FFFu + ((u >> 16) & 1u);   // round-to-nearest-even
    return (unsigned short)(u >> 16);
}

// pack two f32 -> bf16x2 in one instruction where available (v_cvt_pk_bf16_f32)
__device__ __forceinline__ unsigned pack2_bf16(float a, float b) {
#if __has_builtin(__builtin_amdgcn_cvt_pk_bf16_f32)
    typedef __attribute__((ext_vector_type(2))) __bf16 v2bf;
    union { v2bf v; unsigned u; } cv;
    cv.v = __builtin_amdgcn_cvt_pk_bf16_f32(a, b);
    return cv.u;
#else
    return (unsigned)f32_to_bf16(a) | ((unsigned)f32_to_bf16(b) << 16);
#endif
}

// hardware tanh (gfx1250 v_tanh_f32 trans-op; co-executes with matrix pipe)
__device__ __forceinline__ float fast_tanh(float v) {
#if __has_builtin(__builtin_amdgcn_tanhf)
    return __builtin_amdgcn_tanhf(v);
#elif __has_builtin(__builtin_amdgcn_tanh_f32)
    return __builtin_amdgcn_tanh_f32(v);
#else
    float r;
    asm("v_tanh_f32 %0, %1" : "=v"(r) : "v"(v));
    return r;
#endif
}

// branch-free exp via v_exp_f32
__device__ __forceinline__ float fast_exp(float v) {
#if __has_builtin(__builtin_amdgcn_exp2f)
    return __builtin_amdgcn_exp2f(v * 1.4426950408889634f);
#else
    return exp2f(v * 1.4426950408889634f);
#endif
}

// monotone float<->uint mapping so atomicMax(unsigned) orders like float
__device__ __forceinline__ unsigned enc_max(float f) {
    unsigned u = __float_as_uint(f);
    return (u & 0x80000000u) ? ~u : (u | 0x80000000u);
}
__device__ __forceinline__ float dec_max(unsigned u) {
    return __uint_as_float((u & 0x80000000u) ? (u & 0x7FFFFFFFu) : ~u);
}

// ---------------------------------------------------------------------------
// Kernel 1: zero out, init segment buffers, and pre-swizzle W1 (f32 -> bf16)
// into WMMA B-fragment order: 64 fragments (nt,kt) of 32x16, each fragment
// stored as [lane][16 bf16].  B 32x16 bf16 layout used here (A matches): lane<16 -> N=lane,
// K=0..15 ; lane>=16 -> N=lane-16, K=16..31 (2 packed per VGPR).
// ---------------------------------------------------------------------------
__global__ __launch_bounds__(256) void prep_kernel(
    const float* __restrict__ W1, float* __restrict__ out,
    unsigned* __restrict__ seg_max, float* __restrict__ seg_sum,
    unsigned short* __restrict__ w1frag, long out_elems)
{
    long tid = (long)blockIdx.x * 256 + threadIdx.x;
    if (tid < out_elems) out[tid] = 0.0f;
    if (tid < NUM_GRAPHS) {
        seg_max[tid] = enc_max(-__builtin_inff());
        seg_sum[tid] = 0.0f;
    }
    if (tid < 64 * 512) {
        int frag = (int)(tid >> 9);     // nt*8 + kt
        int r    = (int)(tid & 511);
        int lane = r >> 4;
        int i    = r & 15;
        int nt = frag >> 3, kt = frag & 7;
        int n  = nt * 16 + (lane & 15);
        int ko = (lane < 16) ? i : (i + 16);
        int k  = kt * 32 + ko;
        w1frag[tid] = f32_to_bf16(W1[(long)k * HALF_H + n]);
    }
}

// ---------------------------------------------------------------------------
// Kernel 2: gate = tanh(x@W1 + b1) @ W2 + b2 via v_wmma_f32_16x16x32_bf16,
// plus per-segment atomic max of the gate.
// Block = 256 threads = 8 waves; wave handles 16 nodes.
// LDS: 64KB W1 fragments + 8 * 8KB staged bf16 x rows = 128KB dynamic.
// W1 fragments arrive via the async global->LDS engine (ASYNCcnt) when the
// toolchain exposes it; falls back to a plain VGPR-staged copy otherwise.
// ---------------------------------------------------------------------------
__global__ __launch_bounds__(256) void gate_kernel(
    const float* __restrict__ x, const float* __restrict__ b1,
    const float* __restrict__ W2, const float* __restrict__ b2,
    const int* __restrict__ batch, const unsigned short* __restrict__ w1frag,
    float* __restrict__ gate, unsigned* __restrict__ seg_max, int N)
{
    extern __shared__ char smem[];
    unsigned short* s_w1 = (unsigned short*)smem;            // 32768 bf16 = 64KB
    unsigned short* s_x  = (unsigned short*)(smem + 65536);  // 8*16*256 bf16 = 64KB

    int t    = threadIdx.x;
    int wave = t >> 5;
    int lane = t & 31;

    // cooperative copy of fragmented W1 into LDS (4096 * 16B)
#if __has_builtin(__builtin_amdgcn_global_load_async_to_lds_b128) && \
    __has_builtin(__builtin_amdgcn_s_wait_asynccnt)
    {
        typedef int v4i __attribute__((vector_size(16)));
        typedef __attribute__((address_space(1))) v4i gv4i;  // global
        typedef __attribute__((address_space(3))) v4i lv4i;  // LDS
        const char* srcb = (const char*)w1frag;
        char*       dstb = (char*)s_w1;
        #pragma unroll
        for (int j = 0; j < 16; ++j) {
            int boff = (t + 256 * j) * 16;
            __builtin_amdgcn_global_load_async_to_lds_b128(
                (gv4i*)(uintptr_t)(srcb + boff),
                (lv4i*)(unsigned)(uintptr_t)(dstb + boff),
                0, 0);
        }
        __builtin_amdgcn_s_wait_asynccnt(0);
    }
#else
    {
        const uint4* src = (const uint4*)w1frag;
        uint4*       dst = (uint4*)s_w1;
        #pragma unroll
        for (int j = 0; j < 16; ++j) dst[t + 256 * j] = src[t + 256 * j];
    }
#endif

    long base = (long)blockIdx.x * 128 + wave * 16;  // first node of this wave
    unsigned short* xs = s_x + wave * (16 * 256);

    // stage 16 rows of x as bf16 (coalesced float4 loads, 2 per lane per row)
    if (base + 16 <= N) {
        // full tile: no bounds checks
        for (int r = 0; r < 16; ++r) {
            const float4* rp = (const float4*)(x + (base + r) * HIDDEN);
            #pragma unroll
            for (int j = 0; j < 2; ++j) {
                int idx = lane + 32 * j;  // float4 index within row: 0..63
                float4 v = rp[idx];
                uint2 p;
                p.x = pack2_bf16(v.x, v.y);
                p.y = pack2_bf16(v.z, v.w);
                *((uint2*)(xs + r * 256 + idx * 4)) = p;
            }
        }
    } else {
        for (int r = 0; r < 16; ++r) {
            long node = base + r;
            #pragma unroll
            for (int j = 0; j < 2; ++j) {
                int idx = lane + 32 * j;
                float4 v = make_float4(0.f, 0.f, 0.f, 0.f);
                if (node < N) v = ((const float4*)(x + node * HIDDEN))[idx];
                uint2 p;
                p.x = pack2_bf16(v.x, v.y);
                p.y = pack2_bf16(v.z, v.w);
                *((uint2*)(xs + r * 256 + idx * 4)) = p;
            }
        }
    }
    __syncthreads();

    // A fragments: 16x32 bf16.  lane<16: M=lane, K = kt*32 + {0..15};
    // lane>=16: M=lane-16, K = kt*32 + {16..31}.
    union AV { v16bf v; uint4 q[2]; };
    AV a[8];
    int m    = lane & 15;
    int koff = (lane < 16) ? 0 : 16;
    #pragma unroll
    for (int kt = 0; kt < 8; ++kt) {
        const unsigned short* rowp = xs + m * 256 + kt * 32 + koff;
        a[kt].q[0] = *((const uint4*)rowp);
        a[kt].q[1] = *((const uint4*)(rowp + 8));
    }

    float gpart[8];
    #pragma unroll
    for (int r = 0; r < 8; ++r) gpart[r] = 0.0f;

    #pragma unroll
    for (int nt = 0; nt < 8; ++nt) {
        v8f acc = {};
        #pragma unroll
        for (int kt = 0; kt < 8; ++kt) {
            AV b;
            const uint4* bp = (const uint4*)s_w1 + (nt * 8 + kt) * 64 + lane * 2;
            b.q[0] = bp[0];
            b.q[1] = bp[1];
            acc = __builtin_amdgcn_wmma_f32_16x16x32_bf16(
                false, a[kt].v, false, b.v, (short)0, acc, false, false);
        }
        // C layout: lane (&15) = N column; vgpr r = M row (low half M=r, high half M=r+8)
        int   col = nt * 16 + m;
        float b1v = b1[col];
        float w2v = W2[col];
        #pragma unroll
        for (int r = 0; r < 8; ++r) {
            float h = fast_tanh(acc[r] + b1v);
            gpart[r] += h * w2v;
        }
    }

    // reduce the 16 N-columns inside each half-wave
    #pragma unroll
    for (int r = 0; r < 8; ++r) {
        float v = gpart[r];
        v += __shfl_xor(v, 1, 32);
        v += __shfl_xor(v, 2, 32);
        v += __shfl_xor(v, 4, 32);
        v += __shfl_xor(v, 8, 32);
        gpart[r] = v;
    }

    float b2v = b2[0];
    if (lane == 0 || lane == 16) {
        int mbase = (lane == 0) ? 0 : 8;
        #pragma unroll
        for (int r = 0; r < 8; ++r) {
            long node = base + mbase + r;
            if (node < N) {
                float gv = gpart[r] + b2v;
                gate[node] = gv;
                atomicMax(&seg_max[batch[node]], enc_max(gv));
            }
        }
    }
}

// ---------------------------------------------------------------------------
// Kernel 3: g = exp(gate - segmax[batch]); segmented wave scan (batch sorted)
// so only segment-tail lanes issue atomicAdd into seg_sum.
// ---------------------------------------------------------------------------
__global__ __launch_bounds__(256) void softmax_sum_kernel(
    const int* __restrict__ batch, float* __restrict__ gate /* in gate, out g */,
    const unsigned* __restrict__ seg_max, float* __restrict__ seg_sum, int N)
{
    long n   = (long)blockIdx.x * 256 + threadIdx.x;
    int lane = threadIdx.x & 31;

    int   seg = -1;
    float g   = 0.0f;
    if (n < N) {
        seg = batch[n];
        g   = fast_exp(gate[n] - dec_max(seg_max[seg]));
        gate[n] = g;
    }

    float v = g;
    #pragma unroll
    for (int off = 1; off < 32; off <<= 1) {
        float ov = __shfl_up(v, (unsigned)off, 32);
        int   os = __shfl_up(seg, (unsigned)off, 32);
        if (lane >= off && os == seg) v += ov;
    }
    int  nseg = __shfl_down(seg, 1u, 32);
    bool last = (lane == 31) || (nseg != seg);
    if (last && seg >= 0) atomicAdd(&seg_sum[seg], v);
}

// ---------------------------------------------------------------------------
// Kernel 4: out[seg][h] += (g[n]/segsum[seg]) * x[n][h].
// Block = 2048 contiguous nodes, thread = one of 256 columns, accumulator
// flushed only at segment boundaries.
// ---------------------------------------------------------------------------
__global__ __launch_bounds__(256) void readout_kernel(
    const float* __restrict__ x, const int* __restrict__ batch,
    const float* __restrict__ g, const float* __restrict__ seg_sum,
    float* __restrict__ out, int N)
{
    int  t     = threadIdx.x;
    long start = (long)blockIdx.x * NPB;
    long end   = start + NPB;
    if (end > N) end = N;
    if (start >= end) return;

    int   curseg = batch[start];
    float inv    = 1.0f / (seg_sum[curseg] + 1e-16f);
    float acc    = 0.0f;

    for (long n = start; n < end; ++n) {
        __builtin_prefetch(&x[(n + 8) * HIDDEN + t], 0, 0);
        int seg = batch[n];
        if (seg != curseg) {
            atomicAdd(&out[(long)curseg * HIDDEN + t], acc);
            acc    = 0.0f;
            curseg = seg;
            inv    = 1.0f / (seg_sum[seg] + 1e-16f);
        }
        acc += (g[n] * inv) * x[n * HIDDEN + t];
    }
    atomicAdd(&out[(long)curseg * HIDDEN + t], acc);
}

// ---------------------------------------------------------------------------
extern "C" void kernel_launch(void* const* d_in, const int* in_sizes, int n_in,
                              void* d_out, int out_size, void* d_ws, size_t ws_size,
                              hipStream_t stream) {
    const float* x     = (const float*)d_in[0];
    const float* W1    = (const float*)d_in[1];
    const float* b1    = (const float*)d_in[2];
    const float* W2    = (const float*)d_in[3];
    const float* b2    = (const float*)d_in[4];
    const int*   batch = (const int*)d_in[5];
    float*       out   = (float*)d_out;

    int N = in_sizes[0] / HIDDEN;

    // workspace layout
    char* ws = (char*)d_ws;
    float* gate = (float*)ws;                                   // N floats (gate, then g)
    size_t off = (((size_t)N * 4) + 255) & ~(size_t)255;
    unsigned* seg_max = (unsigned*)(ws + off); off += NUM_GRAPHS * 4;
    float*    seg_sum = (float*)(ws + off);    off += NUM_GRAPHS * 4;
    unsigned short* w1frag = (unsigned short*)(ws + off);       // 64*512 bf16 = 64KB

    long out_elems = (long)NUM_GRAPHS * HIDDEN;  // 2,097,152 (covers 32768 & 8192 too)
    {
        int blocks = (int)((out_elems + 255) / 256);
        prep_kernel<<<blocks, 256, 0, stream>>>(W1, out, seg_max, seg_sum, w1frag, out_elems);
    }
    {
        int blocks = (N + 127) / 128;
        gate_kernel<<<blocks, 256, 131072, stream>>>(x, b1, W2, b2, batch, w1frag,
                                                     gate, seg_max, N);
    }
    {
        int blocks = (N + 255) / 256;
        softmax_sum_kernel<<<blocks, 256, 0, stream>>>(batch, gate, seg_max, seg_sum, N);
    }
    {
        int blocks = (N + NPB - 1) / NPB;
        readout_kernel<<<blocks, 256, 0, stream>>>(x, batch, gate, seg_sum, out, N);
    }
}
